// CC_Decoder_30820685316837
// MI455X (gfx1250) — compile-verified, matrix operand order checked by reading, not run on hardware
//
#include <hip/hip_runtime.h>
#include <hip/hip_bf16.h>
#include <math.h>

typedef __attribute__((ext_vector_type(16))) _Float16 v16h;
typedef __attribute__((ext_vector_type(8)))  float    v8f;
typedef __attribute__((ext_vector_type(4)))  unsigned su4;
typedef __attribute__((ext_vector_type(8)))  unsigned su8;

#define C1V    1024
#define NFV    256
#define LLAY   4
#define WDV    1028
#define WDP    1040      // padded to multiple of 16
#define NPIX   16384
#define BATCH  8
#define MT     128       // pixel tile per workgroup in MLP
#define AST    264       // act LDS row stride in halves (bank-conflict-free A loads)

union FragH { v16h h; uint4 q[2]; _Float16 e[16]; };

static __device__ __forceinline__ uint4 ldq(const void* p) { return *(const uint4*)p; }

static __device__ __forceinline__ void ldfrag(FragH& f, const _Float16* base, int kc) {
  const _Float16* p = base + kc * 32;
  f.q[0] = ldq(p);
  f.q[1] = ldq(p + 16);
}

// ---------------------------------------------------------------------------
// Kernel 1: f32 -> f16 packing (conv_w padded, x transposed per batch, wfine)
// ---------------------------------------------------------------------------
__global__ void cvt_inputs(const float* __restrict__ x,
                           const float* __restrict__ conv_w,
                           const float* __restrict__ wfine_w,
                           _Float16* __restrict__ conv_wh,   // [1040][1024]
                           _Float16* __restrict__ xTh,       // [B][256 p][1024 c]
                           _Float16* __restrict__ wfineh)    // [256][256]
{
  const long N1 = (long)WDP * C1V;            // 1,064,960
  const long N2 = (long)BATCH * NFV * C1V;    // 2,097,152
  const long N3 = (long)NFV * NFV;            // 65,536
  long i = (long)blockIdx.x * blockDim.x + threadIdx.x;
  if (i < N1) {
    int row = (int)(i >> 10), col = (int)(i & 1023);
    float v = (row < WDV) ? conv_w[(long)row * C1V + col] : 0.0f;
    conv_wh[i] = (_Float16)v;
  } else if (i < N1 + N2) {
    long j = i - N1;
    int c = (int)(j & 1023);
    int p = (int)((j >> 10) & 255);
    int b = (int)(j >> 18);
    xTh[j] = (_Float16)x[((long)b * C1V + c) * NFV + p];
  } else if (i < N1 + N2 + N3) {
    long j = i - N1 - N2;
    wfineh[j] = (_Float16)wfine_w[j];
  }
}

// ---------------------------------------------------------------------------
// Kernel 2: generic WMMA f16 GEMM, C(M x 256) = A(M x K) * Bcol(K x 256)
//   A row-major (lda), B column-major (per-column K-contiguous, ldk)
//   8 waves, each wave owns 2 N-tiles. Software-pipelined: fragment loads for
//   K-chunk k+1 are issued before the WMMAs of chunk k (ping-pong, unroll x2).
// ---------------------------------------------------------------------------
__global__ void __launch_bounds__(256)
gemm16(const _Float16* __restrict__ A, int lda, long strideA,
       const _Float16* __restrict__ Bc, int ldk, long strideB,
       const float* __restrict__ biasRow, int biasRowMax,
       const float* __restrict__ biasCol,
       int K, _Float16* __restrict__ Cout, long strideC)
{
  const int tid  = threadIdx.x;
  const int wave = tid >> 5, lane = tid & 31;
  const int nl = lane & 15, s = lane >> 4;
  const int mtile = blockIdx.x, b = blockIdx.y;
  const _Float16* Ab = A + strideA * b;
  const _Float16* Bb = Bc + strideB * b;
  _Float16* Cb = Cout + strideC * b;
  const int mrow = mtile * 16 + nl;
  const int n0 = wave * 32 + nl;   // N-tile pair (2*wave, 2*wave+1)
  const int n1 = n0 + 16;
  const _Float16* apb  = Ab + (long)mrow * lda + s * 8;
  const _Float16* bp0b = Bb + (long)n0 * ldk + s * 8;
  const _Float16* bp1b = Bb + (long)n1 * ldk + s * 8;
  v8f acc0 = {}, acc1 = {};
  const int nk = K >> 5;           // 32 or 8 (even)
  FragH aA, b0A, b1A, aB, b0B, b1B;
  ldfrag(aA, apb, 0); ldfrag(b0A, bp0b, 0); ldfrag(b1A, bp1b, 0);
  for (int kc = 0; kc < nk; kc += 2) {
    // prefetch chunk kc+1 while issuing WMMAs for chunk kc
    ldfrag(aB, apb, kc + 1); ldfrag(b0B, bp0b, kc + 1); ldfrag(b1B, bp1b, kc + 1);
    acc0 = __builtin_amdgcn_wmma_f32_16x16x32_f16(false, aA.h, false, b0A.h, (short)0, acc0, false, false);
    acc1 = __builtin_amdgcn_wmma_f32_16x16x32_f16(false, aA.h, false, b1A.h, (short)0, acc1, false, false);
    // prefetch chunk kc+2 (clamped on the final pass) while chunk kc+1 computes
    int k2 = (kc + 2 < nk) ? kc + 2 : kc;
    ldfrag(aA, apb, k2); ldfrag(b0A, bp0b, k2); ldfrag(b1A, bp1b, k2);
    acc0 = __builtin_amdgcn_wmma_f32_16x16x32_f16(false, aB.h, false, b0B.h, (short)0, acc0, false, false);
    acc1 = __builtin_amdgcn_wmma_f32_16x16x32_f16(false, aB.h, false, b1B.h, (short)0, acc1, false, false);
  }
  float bc0 = biasCol ? biasCol[n0] : 0.0f;
  float bc1 = biasCol ? biasCol[n1] : 0.0f;
  #pragma unroll
  for (int v = 0; v < 8; ++v) {
    int row = mtile * 16 + s * 8 + v;
    float br = (biasRow && row < biasRowMax) ? biasRow[row] : 0.0f;
    Cb[(long)row * NFV + n0] = (_Float16)(acc0[v] + br + bc0);
    Cb[(long)row * NFV + n1] = (_Float16)(acc1[v] + br + bc1);
  }
}

// ---------------------------------------------------------------------------
// Kernel 3: repack W rows into WMMA B-fragment-linear layout per (b, layer):
//   Wfrag[b][jl][kc(8)][nt(16)] = 512 halves: [chunk(2)][lane(32)][8 halves]
// ---------------------------------------------------------------------------
__global__ void repack_w(const _Float16* __restrict__ Wh, _Float16* __restrict__ Wfrag)
{
  int g = blockIdx.x * blockDim.x + threadIdx.x;   // 131072 fragment-lanes
  int lane = g & 31;
  int nt = (g >> 5) & 15;
  int kc = (g >> 9) & 7;
  int jl = (g >> 12) & 3;
  int b  = g >> 14;
  int nl = lane & 15, s = lane >> 4;
  int col = nt * 16 + nl;
  const _Float16* Wb = Wh + (long)b * WDP * NFV;
  int rbase = 257 * jl + kc * 32 + s * 8;
  FragH f;
  #pragma unroll
  for (int i = 0; i < 8; ++i) f.e[i]     = Wb[(long)(rbase + i) * NFV + col];
  #pragma unroll
  for (int i = 0; i < 8; ++i) f.e[8 + i] = Wb[(long)(rbase + 16 + i) * NFV + col];
  _Float16* dst = Wfrag + ((long)((b * 4 + jl) * 128 + kc * 16 + nt)) * 512;
  *(uint4*)(dst + lane * 8)       = f.q[0];
  *(uint4*)(dst + 256 + lane * 8) = f.q[1];
}

// ---------------------------------------------------------------------------
// Kernel 4: fused MLP. One WG = 128 pixels x 256 features kept in LDS across
// all 4 layers + final projection. 8 waves; wave owns a 16-pixel strip and
// 16 f32 accumulator tiles. Per-layer weights (128 KB f16) are DMA'd into
// LDS by the Tensor Data Mover; B-fragment LDS reads are pipelined one
// N-tile ahead of the WMMA stream.
// ---------------------------------------------------------------------------
__global__ void __launch_bounds__(256)
mlp_fused(const _Float16* __restrict__ Wh,     // row-major W (bias rows)
          const _Float16* __restrict__ Wfrag,  // fragment-linear weights
          const float* __restrict__ last1_w,
          const float* __restrict__ last1_b,
          const float* __restrict__ prelu_a,
          float* __restrict__ out)
{
  extern __shared__ char smem[];
  _Float16* act    = (_Float16*)smem;                          // 128*264*2 = 67584 B
  _Float16* wbuf   = (_Float16*)(smem + 67584);                // 131072 B
  float*    lw     = (float*)(smem + 67584 + 131072);          // 3072 B
  float*    coeffs = (float*)(smem + 67584 + 131072 + 3072);   // 256 B

  const int tid  = threadIdx.x;
  const int wave = tid >> 5, lane = tid & 31;
  const int nl = lane & 15, s = lane >> 4;
  const int ptile = blockIdx.x, b = blockIdx.y;
  const int pbase = ptile * MT;
  const int mbase = wave * 16;

  if (tid < 64)  coeffs[tid] = 6.28318530717958647f * __expf(2.30258509299404568f * ((float)tid * (1.0f / 64.0f)));
  // async copy of last1_w (3x256 f32) into LDS, tracked by ASYNCcnt
  for (int i = tid; i < 768; i += 256) {
    unsigned ldsoff = (unsigned)(unsigned long)(void*)(lw + i);
    unsigned long gaddr = (unsigned long)(const void*)(last1_w + i);
    asm volatile("global_load_async_to_lds_b32 %0, %1, off"
                 :: "v"(ldsoff), "v"(gaddr) : "memory");
  }
  asm volatile("s_wait_asynccnt 0" ::: "memory");
  __syncthreads();

  // ---- positional-encoding activations (x2) straight into LDS ----
  const float V0 = -0.99999f;
  const float R  = 1.99999f / 256.0f;
  for (int e = tid; e < MT * NFV; e += 256) {
    int row = e >> 8, col = e & 255;
    int p = pbase + row;
    int py = p >> 7, px = p & 127;
    int c = col >> 7, t = (col >> 6) & 1, j = col & 63;
    float coord = V0 + R + 2.0f * R * (float)(c ? px : py);
    float val = coeffs[j] * coord;
    act[row * AST + col] = (_Float16)(t ? __sinf(val) : __cosf(val));
  }

  const float pa = prelu_a[0];
  const _Float16* Whb = Wh + (long)b * WDP * NFV;

  for (int jl = 0; jl < LLAY; ++jl) {
    __syncthreads();   // act ready / prev layer's wbuf reads done
    // ---- stage this layer's weights (128 KB contiguous) via the TDM ----
    if (wave == 0) {
      unsigned long gaddr = (unsigned long)(const void*)Wfrag
                          + (unsigned long)(b * 4 + jl) * 131072UL;
      unsigned ldsaddr = (unsigned)(unsigned long)(void*)wbuf; // LDS byte offset
      su4 g0;
      g0[0] = 1u;                                  // count=1, user descriptor
      g0[1] = ldsaddr;                             // lds_addr (bytes)
      g0[2] = (unsigned)(gaddr & 0xffffffffu);     // global_addr[31:0]
      g0[3] = (unsigned)((gaddr >> 32) & 0x01ffffffu) | (2u << 30); // addr[56:32] | type=2
      su8 g1;
      g1[0] = (3u << 16);        // workgroup_mask=0, data_size=3 (8-byte elements)
      g1[1] = (16384u << 16);    // tensor_dim0[15:0]=16384 in bits[31:16]
      g1[2] = (1u << 16);        // tensor_dim0 hi=0 | tensor_dim1 lo=1
      g1[3] = (16384u << 16);    // tensor_dim1 hi=0 | tile_dim0=16384
      g1[4] = 0u;                // tile_dim1=0 (1D), tile_dim2=0
      g1[5] = 16384u;            // tensor_dim0_stride lo
      g1[6] = 0u;                // stride hi / dim1_stride lo
      g1[7] = 0u;
      asm volatile("tensor_load_to_lds %0, %1" :: "s"(g0), "s"(g1) : "memory");
      __builtin_amdgcn_s_wait_tensorcnt(0);
    }
    if (jl + 1 < LLAY)
      __builtin_prefetch(Wfrag + (long)(b * 4 + jl + 1) * NFV * NFV, 0, 1);
    // init accumulators with the layer bias row (W row 257*jl + 256)
    v8f acc[16];
    {
      const _Float16* brow = Whb + (long)(257 * jl + 256) * NFV;
      #pragma unroll
      for (int nt = 0; nt < 16; ++nt) {
        float bv = (float)brow[nt * 16 + nl];
        #pragma unroll
        for (int v = 0; v < 8; ++v) acc[nt][v] = bv;
      }
    }
    __syncthreads();   // TDM done (wave0 waited TENSORcnt) -> weights visible
    const _Float16* wlane = wbuf + lane * 8;
    for (int kc = 0; kc < 8; ++kc) {
      FragH a;
      const _Float16* ap = act + (mbase + nl) * AST + kc * 32 + s * 8;
      a.q[0] = *(const uint4*)ap;
      a.q[1] = *(const uint4*)(ap + 16);
      // pipelined B-fragment LDS reads: load nt+1 before WMMA(nt)
      const _Float16* bbase = wlane + (kc << 13);   // kc*16 groups * 512 halves
      FragH bfA, bfB;
      bfA.q[0] = *(const uint4*)(bbase);
      bfA.q[1] = *(const uint4*)(bbase + 256);
      #pragma unroll
      for (int nt = 0; nt < 16; nt += 2) {
        const _Float16* bpB = bbase + ((nt + 1) << 9);
        bfB.q[0] = *(const uint4*)(bpB);
        bfB.q[1] = *(const uint4*)(bpB + 256);
        acc[nt] = __builtin_amdgcn_wmma_f32_16x16x32_f16(false, a.h, false, bfA.h, (short)0, acc[nt], false, false);
        if (nt + 2 < 16) {
          const _Float16* bpA = bbase + ((nt + 2) << 9);
          bfA.q[0] = *(const uint4*)(bpA);
          bfA.q[1] = *(const uint4*)(bpA + 256);
        }
        acc[nt + 1] = __builtin_amdgcn_wmma_f32_16x16x32_f16(false, a.h, false, bfB.h, (short)0, acc[nt + 1], false, false);
      }
    }
    __syncthreads();   // all waves done reading act for this layer
    // PReLU + write new activations in place
    #pragma unroll
    for (int nt = 0; nt < 16; ++nt) {
      #pragma unroll
      for (int v = 0; v < 8; ++v) {
        float f = acc[nt][v];
        f = (f >= 0.0f) ? f : pa * f;
        act[(mbase + s * 8 + v) * AST + nt * 16 + nl] = (_Float16)f;
      }
    }
  }
  __syncthreads();

  // ---- final 256 -> 3 projection, SiLU, transposed store ----
  if (tid < MT) {
    float c0 = last1_b[0], c1 = last1_b[1], c2 = last1_b[2];
    const _Float16* arow = act + tid * AST;
    for (int k = 0; k < NFV; ++k) {
      float a = (float)arow[k];
      c0 += a * lw[k];
      c1 += a * lw[256 + k];
      c2 += a * lw[512 + k];
    }
    int p = pbase + tid;
    float s0 = c0 / (1.0f + __expf(-c0));
    float s1 = c1 / (1.0f + __expf(-c1));
    float s2 = c2 / (1.0f + __expf(-c2));
    out[((long)b * 3 + 0) * NPIX + p] = s0;
    out[((long)b * 3 + 1) * NPIX + p] = s1;
    out[((long)b * 3 + 2) * NPIX + p] = s2;
  }
}

// ---------------------------------------------------------------------------
extern "C" void kernel_launch(void* const* d_in, const int* in_sizes, int n_in,
                              void* d_out, int out_size, void* d_ws, size_t ws_size,
                              hipStream_t stream)
{
  (void)in_sizes; (void)n_in; (void)out_size; (void)ws_size;
  const float* x       = (const float*)d_in[0];
  const float* conv_w  = (const float*)d_in[1];
  const float* conv_b  = (const float*)d_in[2];
  const float* wfine_w = (const float*)d_in[3];
  const float* wfine_b = (const float*)d_in[4];
  const float* last1_w = (const float*)d_in[5];
  const float* last1_b = (const float*)d_in[6];
  const float* prelu_a = (const float*)d_in[7];

  char* ws = (char*)d_ws;
  _Float16* conv_wh = (_Float16*)(ws + 0);          // 1040*1024*2      = 2,129,920
  _Float16* xTh     = (_Float16*)(ws + 2129920);    // 8*256*1024*2     = 4,194,304
  _Float16* x1h     = (_Float16*)(ws + 6324224);    // 8*1040*256*2     = 4,259,840
  _Float16* Wh      = (_Float16*)(ws + 10584064);   // 8*1040*256*2     = 4,259,840
  _Float16* Wfrag   = (_Float16*)(ws + 14843904);   // 8*4*64K*2        = 4,194,304
  _Float16* wfineh  = (_Float16*)(ws + 19038208);   // 256*256*2        = 131,072

  // 1) pack inputs to f16
  {
    long total = 1064960L + 2097152L + 65536L;
    int blocks = (int)((total + 255) / 256);
    cvt_inputs<<<blocks, 256, 0, stream>>>(x, conv_w, wfine_w, conv_wh, xTh, wfineh);
  }
  // 2) conv GEMM: x1[b] = conv_w(1040x1024) * x_b^T-col(1024x256) + conv_b (rows)
  gemm16<<<dim3(65, BATCH), 256, 0, stream>>>(
      conv_wh, 1024, 0L,
      xTh, 1024, (long)NFV * C1V,
      conv_b, WDV, nullptr,
      1024, x1h, (long)WDP * NFV);
  // 3) wfine GEMM: W[b] = x1[b](1040x256) * wfine^T-col(256x256) + wfine_b (cols)
  gemm16<<<dim3(65, BATCH), 256, 0, stream>>>(
      x1h, 256, (long)WDP * NFV,
      wfineh, 256, 0L,
      nullptr, 0, wfine_b,
      256, Wh, (long)WDP * NFV);
  // 4) repack per-layer weight matrices into fragment-linear layout
  repack_w<<<512, 256, 0, stream>>>(Wh, Wfrag);
  // 5) fused 4-layer MLP + final projection
  {
    size_t shmem = 67584 + 131072 + 3072 + 256;   // 201,984 B (< 320 KB WGP LDS)
    mlp_fused<<<dim3(128, BATCH), 256, shmem, stream>>>(
        Wh, Wfrag, last1_w, last1_b, prelu_a, (float*)d_out);
  }
}
